// RUMABlock_4398046511443
// MI455X (gfx1250) — matrix operand order repaired and checked by reading, not compiled, not run on hardware
//
#include <hip/hip_runtime.h>
#include <hip/hip_bf16.h>

#define DEV __device__ __forceinline__

typedef __attribute__((ext_vector_type(16))) __bf16 v16bf;
typedef __attribute__((ext_vector_type(8)))  __bf16 v8bf;
typedef __attribute__((ext_vector_type(8)))  float  v8f;
typedef __attribute__((address_space(3)))    __bf16 lds_bf16;

static constexpr int Bn   = 4;
static constexpr int Sn   = 2048;
static constexpr int Dn   = 1024;
static constexpr int Tn   = Bn * Sn;     // 8192 tokens
static constexpr int K2n  = 2 * Dn;      // 2048 (cm feature dim)
static constexpr int NEXPn = 4;

#define CAT16(lo, hi) __builtin_shufflevector((lo), (hi), 0,1,2,3,4,5,6,7,8,9,10,11,12,13,14,15)
#define WMMA_BF16(a, b, c) __builtin_amdgcn_wmma_f32_16x16x32_bf16(false,(a),false,(b),(short)0,(c),false,false)

// Async global->LDS copy of 16 bytes (CDNA5 GLOBAL_LOAD_ASYNC_TO_LDS_B128,
// tracked by ASYNCcnt; bypasses VGPRs entirely).
DEV void async_copy16(__bf16* lds_dst, const __bf16* gsrc) {
  lds_bf16* lp = (lds_bf16*)lds_dst;
  asm volatile("global_load_async_to_lds_b128 %0, %1, off"
               :: "v"(lp), "v"(gsrc)
               : "memory");
}
DEV void wait_async_le5() { asm volatile("s_wait_asynccnt 0x5" ::: "memory"); }
DEV void wait_async_0()   { asm volatile("s_wait_asynccnt 0x0" ::: "memory"); }

DEV __bf16 f2bf(float f) {
  union { float f; unsigned u; } in; in.f = f;
  unsigned u = in.u;
  unsigned r = u + 0x7fffu + ((u >> 16) & 1u);   // round-to-nearest-even
  union { unsigned short s; __bf16 b; } o; o.s = (unsigned short)(r >> 16);
  return o.b;
}
DEV float sigmoidf_(float x) { return 1.f / (1.f + __expf(-x)); }
DEV float siluf_(float x)    { return x * sigmoidf_(x); }

DEV float block_reduce_sum(float v, float* sm) {
  int tid = threadIdx.x;
  sm[tid] = v; __syncthreads();
  for (int s = 128; s > 0; s >>= 1) {
    if (tid < s) sm[tid] += sm[tid + s];
    __syncthreads();
  }
  float r = sm[0]; __syncthreads();
  return r;
}

// ---------------------------------------------------------------------------
// Weight prep: transpose + convert to bf16 so GEMM B-tiles are k-contiguous.
// ---------------------------------------------------------------------------
__global__ void cvt_wq_kernel(const float* __restrict__ Wq, __bf16* __restrict__ WqT) {
  int i = blockIdx.x * 256 + threadIdx.x;        // over Dn*Dn, Wq[k][n]
  int n = i & (Dn - 1);
  int k = i >> 10;
  WqT[(size_t)n * Dn + k] = f2bf(Wq[i]);
}

__global__ void cvt_wg_kernel(const float* __restrict__ Wg, __bf16* __restrict__ WgT) {
  int i = blockIdx.x * 256 + threadIdx.x;        // over 4*2048*1024, Wg[e][f][d]
  int d = i & (Dn - 1);
  int f = (i >> 10) & (K2n - 1);
  int e = i >> 21;
  WgT[((size_t)(e * Dn + d)) * K2n + f] = f2bf(Wg[i]);
}

// ---------------------------------------------------------------------------
// LN1 + bf16 conversions + presence. One block (256 thr) per token.
// ---------------------------------------------------------------------------
__global__ __launch_bounds__(256) void prep_kernel(
    const float* __restrict__ x, const float* __restrict__ mem,
    const float* __restrict__ g1, const float* __restrict__ b1,
    __bf16* __restrict__ normA, __bf16* __restrict__ cmA,
    float* __restrict__ presence) {
  __shared__ float sm[256];
  const int t = blockIdx.x, tid = threadIdx.x;
  const float* xr = x + (size_t)t * Dn;
  const float* mr = mem + (size_t)t * Dn;
  float xv[4], mv[4], s = 0.f, ss = 0.f, ms = 0.f;
#pragma unroll
  for (int i = 0; i < 4; ++i) {
    int idx = tid + i * 256;
    xv[i] = xr[idx]; mv[i] = mr[idx];
    s += xv[i]; ss += xv[i] * xv[i]; ms += mv[i] * mv[i];
  }
  float S  = block_reduce_sum(s, sm);
  float SS = block_reduce_sum(ss, sm);
  float MS = block_reduce_sum(ms, sm);
  float mean = S * (1.f / Dn);
  float var  = SS * (1.f / Dn) - mean * mean;
  float rstd = rsqrtf(var + 1e-5f);
#pragma unroll
  for (int i = 0; i < 4; ++i) {
    int idx = tid + i * 256;
    float nd = (xv[i] - mean) * rstd * g1[idx] + b1[idx];
    normA[(size_t)t * Dn + idx] = f2bf(nd);
    cmA[(size_t)t * K2n + Dn + idx] = f2bf(mv[i]);   // second half of cm = memory
  }
  if (tid == 0) presence[t] = (MS > 1e-12f) ? 1.f : 0.f;
}

// ---------------------------------------------------------------------------
// queries = normA @ WqT^T + bq   (bf16 WMMA, f32 accum)
// BM=64 tokens, BN=256 cols, BK=32. 8 waves: wm=w&3 (M tile), wn=w>>2 (8 N tiles).
// Double-buffered LDS filled by async global->LDS copies (ASYNCcnt pipelined).
// ---------------------------------------------------------------------------
__global__ __launch_bounds__(256) void gemm_q_kernel(
    const __bf16* __restrict__ A, const __bf16* __restrict__ Bt,
    const float* __restrict__ bq, float* __restrict__ Q,
    __bf16* __restrict__ cmA) {
  __shared__ __bf16 As[2][64 * 40];
  __shared__ __bf16 Bs[2][256 * 40];
  const int t0 = blockIdx.x * 64;
  const int n0 = blockIdx.y * 256;
  const int tid = threadIdx.x;
  const int wave = tid >> 5, lane = tid & 31;
  const int wm = wave & 3, wn = wave >> 2;
  const int lrow = lane & 15;
  const int khA = (lane < 16) ? 0 : 8;
  const int khB = (lane < 16) ? 0 : 8;

  v8f acc[8];
#pragma unroll
  for (int j = 0; j < 8; ++j) { v8f z = {}; acc[j] = z; }

  const int a_row = tid >> 2, a_seg = tid & 3;
  const __bf16* a_g = A + (size_t)(t0 + a_row) * Dn + a_seg * 8;
  const __bf16* b_g = Bt + (size_t)(n0 + tid) * Dn;

  auto issue = [&](int buf, int k0) {
    async_copy16(&As[buf][a_row * 40 + a_seg * 8], a_g + k0);
#pragma unroll
    for (int ssg = 0; ssg < 4; ++ssg)
      async_copy16(&Bs[buf][tid * 40 + ssg * 8], b_g + k0 + ssg * 8);
  };

  const int NCH = Dn / 32;
  issue(0, 0);
  for (int i = 0; i < NCH; ++i) {
    const int cur = i & 1;
    if (i + 1 < NCH) { issue(cur ^ 1, (i + 1) * 32); wait_async_le5(); }
    else             { wait_async_0(); }
    __syncthreads();

    v8bf alo = *(const v8bf*)(&As[cur][(wm * 16 + lrow) * 40 + khA]);
    v8bf ahi = *(const v8bf*)(&As[cur][(wm * 16 + lrow) * 40 + khA + 16]);
    v16bf af = CAT16(alo, ahi);
#pragma unroll
    for (int j = 0; j < 8; ++j) {
      const __bf16* bp = &Bs[cur][((wn * 8 + j) * 16 + lrow) * 40 + khB];
      v8bf blo = *(const v8bf*)(bp);
      v8bf bhi = *(const v8bf*)(bp + 16);
      v16bf bf = CAT16(blo, bhi);
      acc[j] = WMMA_BF16(af, bf, acc[j]);
    }
    __syncthreads();   // buffer cur free for reuse at iteration i+1's issue
  }

#pragma unroll
  for (int j = 0; j < 8; ++j) {
    int n = n0 + (wn * 8 + j) * 16 + lrow;
    float bqv = bq[n];
#pragma unroll
    for (int r = 0; r < 8; ++r) {
      int t = t0 + wm * 16 + r + ((lane < 16) ? 0 : 8);
      float q = acc[j][r] + bqv;
      Q[(size_t)t * Dn + n] = q;
      cmA[(size_t)t * K2n + n] = f2bf(q);            // first half of cm = queries
    }
  }
}

// ---------------------------------------------------------------------------
// route_conf = max(softmax(queries @ hyperplanes)). One block per token.
// ---------------------------------------------------------------------------
__global__ __launch_bounds__(256) void route_kernel(
    const float* __restrict__ Q, const float* __restrict__ H, float* __restrict__ rc) {
  __shared__ float sm[16 * 256];
  const int t = blockIdx.x, tid = threadIdx.x;
  float l[16];
#pragma unroll
  for (int n = 0; n < 16; ++n) l[n] = 0.f;
  for (int d = tid; d < Dn; d += 256) {
    float q = Q[(size_t)t * Dn + d];
#pragma unroll
    for (int n = 0; n < 16; ++n) l[n] += q * H[d * 16 + n];
  }
#pragma unroll
  for (int n = 0; n < 16; ++n) sm[n * 256 + tid] = l[n];
  __syncthreads();
  for (int s = 128; s > 0; s >>= 1) {
    if (tid < s)
#pragma unroll
      for (int n = 0; n < 16; ++n) sm[n * 256 + tid] += sm[n * 256 + tid + s];
    __syncthreads();
  }
  if (tid == 0) {
    float mx = sm[0];
#pragma unroll
    for (int n = 1; n < 16; ++n) mx = fmaxf(mx, sm[n * 256]);
    float sum = 0.f;
#pragma unroll
    for (int n = 0; n < 16; ++n) sum += __expf(sm[n * 256] - mx);
    rc[t] = 1.f / sum;   // max softmax prob
  }
}

// ---------------------------------------------------------------------------
// Per-token scalar scoring: tiny MLPs, selectivity, top-2 expert weights.
// ---------------------------------------------------------------------------
__global__ void score_kernel(
    const float* suff, const float* conf, const float* lin, const float* tsc,
    const float* tsco, const float* alg, const float* stale,
    const float* rc, const float* pres,
    const float* Ws1, const float* bs1, const float* Ws2, const float* bs2,
    const float* Ws3, const float* bs3,
    const float* Wr1, const float* br1, const float* Wr2, const float* br2,
    float* __restrict__ ew, float* __restrict__ sel) {
  int t = blockIdx.x * blockDim.x + threadIdx.x;
  if (t >= Tn) return;
  float f[9] = { suff[t], conf[t], lin[t], tsc[t], fmaxf(tsco[t], 0.f),
                 alg[t], stale[t], rc[t], pres[t] };
  float prior = 2.f * f[0] + 1.25f * f[2] + 0.75f * f[5] + 0.35f * f[3] +
                0.25f * f[4] - 1.5f * f[6] - 1.0f * f[1] - 0.5f * (1.f - f[7]);
  float h1[32], h2[32];
  for (int j = 0; j < 32; ++j) {
    float a = bs1[j];
    for (int i = 0; i < 9; ++i) a += f[i] * Ws1[i * 32 + j];
    h1[j] = siluf_(a);
  }
  for (int j = 0; j < 32; ++j) {
    float a = bs2[j];
    for (int i = 0; i < 32; ++i) a += h1[i] * Ws2[i * 32 + j];
    h2[j] = siluf_(a);
  }
  float learned = bs3[0];
  for (int i = 0; i < 32; ++i) learned += h2[i] * Ws3[i];
  sel[t] = sigmoidf_(prior + 0.5f * learned) * f[8];

  float hr[32];
  for (int j = 0; j < 32; ++j) {
    float a = br1[j];
    for (int i = 0; i < 8; ++i) a += f[i] * Wr1[i * 32 + j];
    hr[j] = siluf_(a);
  }
  float lg[4];
  for (int e = 0; e < 4; ++e) {
    float a = br2[e];
    for (int i = 0; i < 32; ++i) a += hr[i] * Wr2[i * 4 + e];
    lg[e] = a;
  }
  float mx = fmaxf(fmaxf(lg[0], lg[1]), fmaxf(lg[2], lg[3]));
  float p[4], psum = 0.f;
  for (int e = 0; e < 4; ++e) { p[e] = __expf(lg[e] - mx); psum += p[e]; }
  for (int e = 0; e < 4; ++e) p[e] /= psum;
  int i0 = 0;
  for (int e = 1; e < 4; ++e) if (p[e] > p[i0]) i0 = e;
  int i1 = (i0 == 0) ? 1 : 0;
  for (int e = 0; e < 4; ++e) if (e != i0 && p[e] > p[i1]) i1 = e;
  float s2 = fmaxf(p[i0] + p[i1], 1e-8f);
  for (int e = 0; e < 4; ++e)
    ew[(size_t)t * 4 + e] = ((e == i0) ? p[i0] : (e == i1) ? p[i1] : 0.f) / s2;
}

// ---------------------------------------------------------------------------
// Fused expert GEMM + full epilogue:
//   logits[t, e, d] = cm[t,:] . WgT[e, d, :]   (bf16 WMMA, K=2048)
//   gates -> cores -> expert transforms -> ew-weighted sum -> y = x + sel*fused
// Wave owns all 4 experts for its (16 tok x 16 d) tiles so the expert
// reduction stays in-register. Async double-buffered LDS pipeline.
// ---------------------------------------------------------------------------
__global__ __launch_bounds__(256) void gemm_fuse_kernel(
    const __bf16* __restrict__ A, const __bf16* __restrict__ Bt,
    const float* __restrict__ bg, const float* __restrict__ Q,
    const float* __restrict__ mem, const float* __restrict__ x,
    const float* __restrict__ ew, const float* __restrict__ sel,
    float* __restrict__ y) {
  __shared__ __bf16 As[2][64 * 40];
  __shared__ __bf16 Bs[2][256 * 40];
  const int t0 = blockIdx.x * 64;
  const int d0 = blockIdx.y * 64;
  const int tid = threadIdx.x;
  const int wave = tid >> 5, lane = tid & 31;
  const int wm = wave & 3;        // token subtile
  const int wd = wave >> 2;       // 0..1 -> d-tiles wd*2 + {0,1}
  const int lrow = lane & 15;
  const int khA = (lane < 16) ? 0 : 8;
  const int khB = (lane < 16) ? 0 : 8;

  v8f acc[2][4];
#pragma unroll
  for (int a = 0; a < 2; ++a)
#pragma unroll
    for (int e = 0; e < 4; ++e) { v8f z = {}; acc[a][e] = z; }

  const int a_row = tid >> 2, a_seg = tid & 3;
  const int b_e = tid >> 6, b_d = tid & 63;
  const __bf16* a_g = A + (size_t)(t0 + a_row) * K2n + a_seg * 8;
  const __bf16* b_g = Bt + (size_t)(b_e * Dn + d0 + b_d) * K2n;

  auto issue = [&](int buf, int k0) {
    async_copy16(&As[buf][a_row * 40 + a_seg * 8], a_g + k0);
#pragma unroll
    for (int ssg = 0; ssg < 4; ++ssg)
      async_copy16(&Bs[buf][tid * 40 + ssg * 8], b_g + k0 + ssg * 8);
  };

  const int NCH = K2n / 32;
  issue(0, 0);
  for (int i = 0; i < NCH; ++i) {
    const int cur = i & 1;
    if (i + 1 < NCH) { issue(cur ^ 1, (i + 1) * 32); wait_async_le5(); }
    else             { wait_async_0(); }
    __syncthreads();

    v8bf alo = *(const v8bf*)(&As[cur][(wm * 16 + lrow) * 40 + khA]);
    v8bf ahi = *(const v8bf*)(&As[cur][(wm * 16 + lrow) * 40 + khA + 16]);
    v16bf af = CAT16(alo, ahi);
#pragma unroll
    for (int dt = 0; dt < 2; ++dt)
#pragma unroll
      for (int e = 0; e < 4; ++e) {
        const __bf16* bp = &Bs[cur][(e * 64 + (wd * 2 + dt) * 16 + lrow) * 40 + khB];
        v8bf blo = *(const v8bf*)(bp);
        v8bf bhi = *(const v8bf*)(bp + 16);
        v16bf bf = CAT16(blo, bhi);
        acc[dt][e] = WMMA_BF16(af, bf, acc[dt][e]);
      }
    __syncthreads();   // buffer cur free for reuse at iteration i+1's issue
  }

#pragma unroll
  for (int dt = 0; dt < 2; ++dt) {
    int d = d0 + (wd * 2 + dt) * 16 + lrow;
    float bg0 = bg[0 * Dn + d], bg1 = bg[1 * Dn + d];
    float bg2 = bg[2 * Dn + d], bg3 = bg[3 * Dn + d];
#pragma unroll
    for (int r = 0; r < 8; ++r) {
      int t = t0 + wm * 16 + r + ((lane < 16) ? 0 : 8);
      size_t idx = (size_t)t * Dn + d;
      float c = Q[idx], m = mem[idx], xv = x[idx];
      float g0 = sigmoidf_(acc[dt][0][r] + bg0);
      float g1 = sigmoidf_(acc[dt][1][r] + bg1);
      float g2 = sigmoidf_(acc[dt][2][r] + bg2);
      float g3 = sigmoidf_(acc[dt][3][r] + bg3);
      float core0 = g0 * c + (1.f - g0) * m;
      float core1 = g1 * c + (1.f - g1) * m;
      float core2 = g2 * c + (1.f - g2) * m;
      float core3 = g3 * c + (1.f - g3) * m;
      float cons   = c + 0.4f  * (core0 - c);
      float base   = core1;
      float bridge = c + 0.85f * (core2 - c) + 0.15f * m;
      float dom    = c + 1.15f * (core3 - c);
      const float* e4 = ew + (size_t)t * 4;
      float fused = e4[0] * cons + e4[1] * base + e4[2] * bridge + e4[3] * dom;
      y[idx] = xv + sel[t] * fused;
    }
  }
}

// ---------------------------------------------------------------------------
// Final LayerNorm. One block per token.
// ---------------------------------------------------------------------------
__global__ __launch_bounds__(256) void ln2_kernel(
    const float* __restrict__ y, const float* __restrict__ g2,
    const float* __restrict__ b2, float* __restrict__ out) {
  __shared__ float sm[256];
  const int t = blockIdx.x, tid = threadIdx.x;
  const float* yr = y + (size_t)t * Dn;
  float v[4], s = 0.f, ss = 0.f;
#pragma unroll
  for (int i = 0; i < 4; ++i) {
    v[i] = yr[tid + i * 256];
    s += v[i]; ss += v[i] * v[i];
  }
  float S  = block_reduce_sum(s, sm);
  float SS = block_reduce_sum(ss, sm);
  float mean = S * (1.f / Dn);
  float var  = SS * (1.f / Dn) - mean * mean;
  float rstd = rsqrtf(var + 1e-5f);
#pragma unroll
  for (int i = 0; i < 4; ++i) {
    int idx = tid + i * 256;
    out[(size_t)t * Dn + idx] = (v[i] - mean) * rstd * g2[idx] + b2[idx];
  }
}

// ---------------------------------------------------------------------------
extern "C" void kernel_launch(void* const* d_in, const int* in_sizes, int n_in,
                              void* d_out, int out_size, void* d_ws, size_t ws_size,
                              hipStream_t stream) {
  const float* x     = (const float*)d_in[0];
  const float* mem   = (const float*)d_in[1];
  const float* suff  = (const float*)d_in[2];
  const float* conf  = (const float*)d_in[3];
  const float* lin   = (const float*)d_in[4];
  const float* tsc   = (const float*)d_in[5];
  const float* tsco  = (const float*)d_in[6];
  const float* alg   = (const float*)d_in[7];
  const float* stale = (const float*)d_in[8];
  const float* g1    = (const float*)d_in[9];
  const float* b1    = (const float*)d_in[10];
  const float* Wq    = (const float*)d_in[11];
  const float* bq    = (const float*)d_in[12];
  const float* H     = (const float*)d_in[13];
  const float* Ws1   = (const float*)d_in[14];
  const float* bs1   = (const float*)d_in[15];
  const float* Ws2   = (const float*)d_in[16];
  const float* bs2   = (const float*)d_in[17];
  const float* Ws3   = (const float*)d_in[18];
  const float* bs3   = (const float*)d_in[19];
  const float* Wr1   = (const float*)d_in[20];
  const float* br1   = (const float*)d_in[21];
  const float* Wr2   = (const float*)d_in[22];
  const float* br2   = (const float*)d_in[23];
  const float* Wg    = (const float*)d_in[24];
  const float* bg    = (const float*)d_in[25];
  const float* g2    = (const float*)d_in[26];
  const float* b2    = (const float*)d_in[27];

  char* ws = (char*)d_ws;
  size_t off = 0;
  auto alloc = [&](size_t bytes) -> char* {
    off = (off + 255) & ~(size_t)255;
    char* p = ws + off;
    off += bytes;
    return p;
  };
  __bf16* normA = (__bf16*)alloc((size_t)Tn * Dn * 2);
  __bf16* cmA   = (__bf16*)alloc((size_t)Tn * K2n * 2);
  __bf16* WqT   = (__bf16*)alloc((size_t)Dn * Dn * 2);
  __bf16* WgT   = (__bf16*)alloc((size_t)NEXPn * Dn * K2n * 2);
  float*  Qf    = (float*)alloc((size_t)Tn * Dn * 4);
  float*  pres  = (float*)alloc((size_t)Tn * 4);
  float*  rc    = (float*)alloc((size_t)Tn * 4);
  float*  ewp   = (float*)alloc((size_t)Tn * 4 * 4);
  float*  selp  = (float*)alloc((size_t)Tn * 4);
  float*  yws   = (float*)alloc((size_t)Tn * Dn * 4);

  cvt_wq_kernel<<<(Dn * Dn) / 256, 256, 0, stream>>>(Wq, WqT);
  cvt_wg_kernel<<<(NEXPn * K2n * Dn) / 256, 256, 0, stream>>>(Wg, WgT);
  prep_kernel<<<Tn, 256, 0, stream>>>(x, mem, g1, b1, normA, cmA, pres);
  gemm_q_kernel<<<dim3(Tn / 64, Dn / 256), 256, 0, stream>>>(normA, WqT, bq, Qf, cmA);
  route_kernel<<<Tn, 256, 0, stream>>>(Qf, H, rc);
  score_kernel<<<(Tn + 255) / 256, 256, 0, stream>>>(
      suff, conf, lin, tsc, tsco, alg, stale, rc, pres,
      Ws1, bs1, Ws2, bs2, Ws3, bs3, Wr1, br1, Wr2, br2, ewp, selp);
  gemm_fuse_kernel<<<dim3(Tn / 64, Dn / 64), 256, 0, stream>>>(
      cmA, WgT, bg, Qf, mem, x, ewp, selp, yws);
  ln2_kernel<<<Tn, 256, 0, stream>>>(yws, g2, b2, (float*)d_out);

  (void)in_sizes; (void)n_in; (void)out_size; (void)ws_size;
}
